// LlamaDecoderBlock_89189290868855
// MI455X (gfx1250) — matrix operand order repaired and checked by reading, not compiled, not run on hardware
//
#include <hip/hip_runtime.h>

typedef __attribute__((ext_vector_type(16))) __bf16 v16bf;
typedef __attribute__((ext_vector_type(8)))  float  v8f;

#define DM   1024
#define D3   3072
#define DFF  4096
#define NH   16
#define HD   64
#define BB   2
#define TT   2048
#define BT   (BB*TT)
#define EPSF 1e-6f

#if __has_include(<hip/amd_detail/amd_gfx1250_TDM.h>)
#define TDM_6ARG 1
#else
#define TDM_6ARG 0
#endif

#if defined(__AMDGCN__) && __has_builtin(__builtin_amdgcn_tensor_load_to_lds)
#define USE_TDM 1
#else
#define USE_TDM 0
#endif

#if USE_TDM
typedef __attribute__((ext_vector_type(4))) unsigned int u32x4;
typedef __attribute__((ext_vector_type(8))) int          i32x8;
typedef __attribute__((ext_vector_type(4))) int          i32x4;

// 2D tile (tile_w elems x tile_h rows, 2-byte elems) global -> LDS via TDM.
// pad_i / pad_a encode D#.pad_interval / D#.pad_amount (LDS row padding).
static __device__ __forceinline__ void tdm_load_2d(
    const void* gaddr, unsigned lds_off, int tile_w, int tile_h,
    long row_stride_elems, int pad_i, int pad_a) {
  unsigned long long ga = (unsigned long long)(uintptr_t)gaddr;
  u32x4 g0;
  g0.x = 1u;                                        // count=1, user descriptor
  g0.y = lds_off;                                   // lds_addr (bytes)
  g0.z = (unsigned)(ga & 0xFFFFFFFFu);              // global_addr[31:0]
  g0.w = (unsigned)((ga >> 32) & 0x1FFFFFFu)        // global_addr[56:32]
       | (2u << 30);                                // type = 2 ("image")
  const unsigned td0 = 0x40000000u, td1 = 0x40000000u;  // huge dims: no OOB
  i32x8 g1;
  g1[0] = (int)((1u << 16)                          // data_size = 2 bytes
              | (1u << 20)                          // pad_enable
              | ((unsigned)pad_i << 22)             // pad_interval
              | ((unsigned)pad_a << 25));           // pad_amount
  g1[1] = (int)((td0 & 0xFFFFu) << 16);             // tensor_dim0[15:0]
  g1[2] = (int)((td0 >> 16) | ((td1 & 0xFFFFu) << 16));
  g1[3] = (int)((td1 >> 16) | ((unsigned)tile_w << 16));   // tile_dim0
  g1[4] = (int)((unsigned)tile_h & 0xFFFFu);        // tile_dim1 (tile_dim2=0)
  unsigned long long st = (unsigned long long)row_stride_elems;
  g1[5] = (int)(st & 0xFFFFFFFFu);                  // tensor_dim0_stride lo
  g1[6] = (int)((st >> 32) & 0xFFFFu);              // stride hi (dim1_stride=0)
  g1[7] = 0;
  i32x4 gz = {0, 0, 0, 0};
#if TDM_6ARG
  i32x8 gz8 = {0, 0, 0, 0, 0, 0, 0, 0};
  __builtin_amdgcn_tensor_load_to_lds(g0, g1, gz, gz, gz8, 0);
#else
  __builtin_amdgcn_tensor_load_to_lds(g0, g1, gz, gz, 0);
#endif
}
#endif // USE_TDM

static __device__ __forceinline__ v16bf ld16(const __bf16* p0, const __bf16* p1) {
  v16bf v;
  ((uint4*)&v)[0] = *(const uint4*)p0;
  ((uint4*)&v)[1] = *(const uint4*)p1;
  return v;
}

// ---------------------------------------------------------------- elementwise
__global__ void cvt_f32_bf16(const float* __restrict__ in, __bf16* __restrict__ out, int n) {
  int i = blockIdx.x * 256 + threadIdx.x;
  if (i < n) out[i] = (__bf16)in[i];
}

__global__ __launch_bounds__(256)
void rmsnorm_bf16(const float* __restrict__ x, const float* __restrict__ wgt,
                  __bf16* __restrict__ out) {
  __shared__ float red[256];
  const int row = blockIdx.x, tid = threadIdx.x;
  const float* xr = x + (size_t)row * DM;
  float vals[4];
  float ss = 0.f;
#pragma unroll
  for (int j = 0; j < 4; ++j) { vals[j] = xr[tid + j * 256]; ss += vals[j] * vals[j]; }
  red[tid] = ss; __syncthreads();
  for (int s = 128; s > 0; s >>= 1) { if (tid < s) red[tid] += red[tid + s]; __syncthreads(); }
  float scale = rsqrtf(red[0] / (float)DM + EPSF);
#pragma unroll
  for (int j = 0; j < 4; ++j) {
    int c = tid + j * 256;
    out[(size_t)row * DM + c] = (__bf16)(vals[j] * scale * wgt[c]);
  }
}

__global__ __launch_bounds__(256)
void rope_split(const float* __restrict__ qkv, __bf16* __restrict__ qb,
                __bf16* __restrict__ kb, __bf16* __restrict__ vb) {
  const int row = blockIdx.x;           // b*TT + t
  const int t = row % TT, b = row / TT;
  const int c0 = threadIdx.x * 4;       // 4 contiguous cols = 2 rope pairs
  const int h = c0 / HD, d0 = c0 % HD;
  const float* r = qkv + (size_t)row * D3;

  const float lb = __logf(10000.0f);
  float a0 = (float)t * __expf(-((float)d0 / (float)HD) * lb);
  float a1 = (float)t * __expf(-((float)(d0 + 2) / (float)HD) * lb);
  float c0f = __cosf(a0), s0f = __sinf(a0);
  float c1f = __cosf(a1), s1f = __sinf(a1);

  size_t ob = (((size_t)(b * NH + h)) * TT + t) * HD + d0;   // [B,H,T,D]

  float q0 = r[c0], q1 = r[c0 + 1], q2 = r[c0 + 2], q3 = r[c0 + 3];
  qb[ob]     = (__bf16)(q0 * c0f - q1 * s0f);
  qb[ob + 1] = (__bf16)(q0 * s0f + q1 * c0f);
  qb[ob + 2] = (__bf16)(q2 * c1f - q3 * s1f);
  qb[ob + 3] = (__bf16)(q2 * s1f + q3 * c1f);

  float k0 = r[DM + c0], k1 = r[DM + c0 + 1], k2 = r[DM + c0 + 2], k3 = r[DM + c0 + 3];
  kb[ob]     = (__bf16)(k0 * c0f - k1 * s0f);
  kb[ob + 1] = (__bf16)(k0 * s0f + k1 * c0f);
  kb[ob + 2] = (__bf16)(k2 * c1f - k3 * s1f);
  kb[ob + 3] = (__bf16)(k2 * s1f + k3 * c1f);

  vb[ob]     = (__bf16)r[2 * DM + c0];
  vb[ob + 1] = (__bf16)r[2 * DM + c0 + 1];
  vb[ob + 2] = (__bf16)r[2 * DM + c0 + 2];
  vb[ob + 3] = (__bf16)r[2 * DM + c0 + 3];
}

__global__ void silu_mul(const __bf16* __restrict__ g, const __bf16* __restrict__ u,
                         __bf16* __restrict__ out) {
  int i0 = (blockIdx.x * 256 + threadIdx.x) * 4;
#pragma unroll
  for (int j = 0; j < 4; ++j) {
    float gv = (float)g[i0 + j];
    float uv = (float)u[i0 + j];
    out[i0 + j] = (__bf16)(gv / (1.f + __expf(-gv)) * uv);
  }
}

// ---------------------------------------------------------------- WMMA GEMM
// C[M,N] = A[M,K] @ B[K,N] (+bias) (+residual), A/B bf16 row-major.
// Workgroup tile 128x128, 8 waves (4m x 2n), wave tile 32x64.
// K-step 32, double-buffered LDS. A tiles staged by the Tensor Data Mover
// (row padding done by the TDM descriptor), B tiles transposed via VGPRs.
template<bool HAS_BIAS, bool HAS_RES, bool OUT_BF16>
__global__ __launch_bounds__(256)
void wmma_gemm(const __bf16* __restrict__ A, const __bf16* __restrict__ B,
               const float* __restrict__ bias, const float* __restrict__ resid,
               void* __restrict__ C, int N, int K) {
  __shared__ __bf16 lds_a[2][128 * 40];   // row-major, pad to 40 elems/row
  __shared__ __bf16 lds_bt[2][128 * 40];  // B transposed: [n][k]

  const int tid = threadIdx.x;
  const int lane = tid & 31, w = tid >> 5;
  const int lane15 = lane & 15, half = lane >> 4;
  const int wm = w & 3, wn = w >> 2;
  const int rowBase = blockIdx.y * 128;
  const int colBase = blockIdx.x * 128;

  const int ar = tid >> 1, ap = (tid & 1) * 16;   // A loader: row, 16-elem part
  const int bk = tid >> 3, bn = (tid & 7) * 16;   // B loader: k row, 16-col group

  v8f acc[2][4] = {};

  uint4 ra0, ra1, rb0, rb1;
  auto loadG = [&](int k0) {
#if !USE_TDM
    const __bf16* ga = A + (size_t)(rowBase + ar) * K + k0 + ap;
    ra0 = ((const uint4*)ga)[0];
    ra1 = ((const uint4*)ga)[1];
#endif
    const __bf16* gb = B + (size_t)(k0 + bk) * N + colBase + bn;
    rb0 = ((const uint4*)gb)[0];
    rb1 = ((const uint4*)gb)[1];
  };
  auto storeL = [&](int buf) {
#if !USE_TDM
    *(uint4*)(lds_a[buf] + ar * 40 + ap)     = ra0;
    *(uint4*)(lds_a[buf] + ar * 40 + ap + 8) = ra1;
#endif
    __bf16 tmp[16];
    *(uint4*)tmp       = rb0;
    *(uint4*)(tmp + 8) = rb1;
#pragma unroll
    for (int i = 0; i < 16; ++i) lds_bt[buf][(bn + i) * 40 + bk] = tmp[i];
  };
#if USE_TDM
  auto tdmA = [&](int k0, int buf) {   // 128 rows x 32 elems, pad 4dw/16dw
    tdm_load_2d(A + (size_t)rowBase * K + k0,
                (unsigned)(uintptr_t)&lds_a[buf][0], 32, 128, (long)K, 3, 3);
  };
  if (w == 0) tdmA(0, 0);
#endif
  loadG(0);
  storeL(0);
#if USE_TDM
  if (w == 0) __builtin_amdgcn_s_wait_tensorcnt(0);
#endif
  __syncthreads();

  const int nk = K / 32;
  for (int ki = 0; ki < nk; ++ki) {
    const int cur = ki & 1, nxt = cur ^ 1;
    if (ki + 1 < nk) {
#if USE_TDM
      if (w == 0) tdmA((ki + 1) * 32, nxt);
#endif
      loadG((ki + 1) * 32);
    }
    if (ki + 2 < nk)   // pull the K-step-after-next into L2 early
      __builtin_prefetch(B + (size_t)((ki + 2) * 32 + bk) * N + colBase + bn, 0, 1);

    v16bf av[2], bv[4];
#pragma unroll
    for (int mi = 0; mi < 2; ++mi) {
      const __bf16* p = lds_a[cur] + (wm * 32 + mi * 16 + lane15) * 40;
      av[mi] = ld16(p + half * 16, p + half * 16 + 8);
    }
#pragma unroll
    for (int ni = 0; ni < 4; ++ni) {
      const __bf16* p = lds_bt[cur] + (wn * 64 + ni * 16 + lane15) * 40 + half * 16;
      bv[ni] = ld16(p, p + 8);
    }
#pragma unroll
    for (int mi = 0; mi < 2; ++mi)
#pragma unroll
      for (int ni = 0; ni < 4; ++ni)
        acc[mi][ni] = __builtin_amdgcn_wmma_f32_16x16x32_bf16(
            false, av[mi], false, bv[ni], (short)0, acc[mi][ni], false, false);

    if (ki + 1 < nk) storeL(nxt);
#if USE_TDM
    if (w == 0) __builtin_amdgcn_s_wait_tensorcnt(0);
#endif
    __syncthreads();
  }

#pragma unroll
  for (int mi = 0; mi < 2; ++mi) {
#pragma unroll
    for (int ni = 0; ni < 4; ++ni) {
      const int col = colBase + wn * 64 + ni * 16 + lane15;
      const float bcol = HAS_BIAS ? bias[col] : 0.f;
#pragma unroll
      for (int i = 0; i < 8; ++i) {
        const int row = rowBase + wm * 32 + mi * 16 + i + 8 * half;
        const size_t idx = (size_t)row * N + col;
        float v = acc[mi][ni][i] + bcol;
        if (HAS_RES) v += resid[idx];
        if (OUT_BF16) ((__bf16*)C)[idx] = (__bf16)v;
        else          ((float*)C)[idx]  = v;
      }
    }
  }
}

// ---------------------------------------------------------------- attention
// Causal flash attention. grid = (T/64, B*H), 128 threads (4 waves),
// wave w owns query rows [qbase + 16w, +16). Key blocks of 32, double-buffered.
// K tiles staged by TDM; V tiles transposed via VGPRs.
__global__ __launch_bounds__(128)
void flash_attn(const __bf16* __restrict__ Q, const __bf16* __restrict__ Kb,
                const __bf16* __restrict__ Vb, __bf16* __restrict__ O) {
  __shared__ __bf16 lds_q[64 * 72];
  __shared__ __bf16 lds_k[2][32 * 72];
  __shared__ __bf16 lds_vt[2][64 * 40];  // V transposed: [d][s]
  __shared__ __bf16 lds_p[64 * 40];      // per-wave 16x32 P patch

  const int tid = threadIdx.x;
  const int lane = tid & 31, w = tid >> 5;
  const int lane15 = lane & 15, half = lane >> 4;
  const int qbase = blockIdx.x * 64;
  const int bh = blockIdx.y;
  const int b = bh / NH, h = bh % NH;

  const int kr = tid >> 2, kp = (tid & 3) * 16;   // V loader: row, 16-elem part

  uint4 rk0, rk1, rv0, rv1;
  auto loadKV = [&](int j0) {
#if !USE_TDM
    const __bf16* gk = Kb + ((size_t)bh * TT + j0 + kr) * HD + kp;
    rk0 = ((const uint4*)gk)[0];
    rk1 = ((const uint4*)gk)[1];
#endif
    const __bf16* gv = Vb + ((size_t)bh * TT + j0 + kr) * HD + kp;
    rv0 = ((const uint4*)gv)[0];
    rv1 = ((const uint4*)gv)[1];
  };
  auto storeKV = [&](int buf) {
#if !USE_TDM
    *(uint4*)(lds_k[buf] + kr * 72 + kp)     = rk0;
    *(uint4*)(lds_k[buf] + kr * 72 + kp + 8) = rk1;
#endif
    __bf16 tmp[16];
    *(uint4*)tmp       = rv0;
    *(uint4*)(tmp + 8) = rv1;
#pragma unroll
    for (int i = 0; i < 16; ++i) lds_vt[buf][(kp + i) * 40 + kr] = tmp[i];
  };
#if USE_TDM
  auto tdmK = [&](int j0, int buf) {   // 32 rows x 64 elems, pad 4dw/32dw
    tdm_load_2d(Kb + ((size_t)bh * TT + j0) * HD,
                (unsigned)(uintptr_t)&lds_k[buf][0], 64, 32, (long)HD, 4, 3);
  };
  if (w == 0) tdmK(0, 0);
#endif

  { // Q tile 64x64
    int r = tid >> 1, part = (tid & 1) * 32;
    const __bf16* g = Q + ((size_t)bh * TT + qbase + r) * HD + part;
#pragma unroll
    for (int j = 0; j < 4; ++j)
      *(uint4*)(lds_q + r * 72 + part + j * 8) = ((const uint4*)g)[j];
  }
  loadKV(0);
  storeKV(0);
#if USE_TDM
  if (w == 0) __builtin_amdgcn_s_wait_tensorcnt(0);
#endif
  __syncthreads();

  v8f o[4] = {};
  float rmax[8], rsum[8];
#pragma unroll
  for (int i = 0; i < 8; ++i) { rmax[i] = -1e30f; rsum[i] = 0.f; }

  const int jend = qbase + 64;
  const int nj = jend / 32;
  for (int ji = 0; ji < nj; ++ji) {
    const int j0 = ji * 32;
    const int cur = ji & 1, nxt = cur ^ 1;
    if (ji + 1 < nj) {
#if USE_TDM
      if (w == 0) tdmK(j0 + 32, nxt);
#endif
      loadKV(j0 + 32);
    }

    // S = Q K^T : 16x32 per wave, K-dim 64 in two steps
    v8f s[2] = {};
#pragma unroll
    for (int ni = 0; ni < 2; ++ni) {
#pragma unroll
      for (int kk = 0; kk < 64; kk += 32) {
        const __bf16* apt = lds_q + (w * 16 + lane15) * 72 + kk;
        v16bf a = ld16(apt + half * 16, apt + half * 16 + 8);
        const __bf16* bpt = lds_k[cur] + (ni * 16 + lane15) * 72 + kk + half * 16;
        v16bf bb = ld16(bpt, bpt + 8);
        s[ni] = __builtin_amdgcn_wmma_f32_16x16x32_bf16(
            false, a, false, bb, (short)0, s[ni], false, false);
      }
    }

    // scale + causal mask
    float sv[2][8];
#pragma unroll
    for (int ni = 0; ni < 2; ++ni) {
      int key = j0 + ni * 16 + lane15;
#pragma unroll
      for (int i = 0; i < 8; ++i) {
        int qr = qbase + w * 16 + i + 8 * half;
        float v = s[ni][i] * 0.125f;   // 1/sqrt(64)
        sv[ni][i] = (key <= qr) ? v : -1e30f;
      }
    }

    // online softmax, rows reduced across the 16 lanes of each half
#pragma unroll
    for (int i = 0; i < 8; ++i) {
      float mx = fmaxf(sv[0][i], sv[1][i]);
#pragma unroll
      for (int d = 8; d >= 1; d >>= 1) mx = fmaxf(mx, __shfl_xor(mx, d, 16));
      float nm = fmaxf(rmax[i], mx);
      float f  = __expf(rmax[i] - nm);
      float p0 = __expf(sv[0][i] - nm);
      float p1 = __expf(sv[1][i] - nm);
      float rs = p0 + p1;
#pragma unroll
      for (int d = 8; d >= 1; d >>= 1) rs += __shfl_xor(rs, d, 16);
      rsum[i] = rsum[i] * f + rs;
      rmax[i] = nm;
#pragma unroll
      for (int nt = 0; nt < 4; ++nt) o[nt][i] *= f;
      int m = w * 16 + i + 8 * half;                 // C-layout -> LDS
      lds_p[m * 40 + lane15]      = (__bf16)p0;
      lds_p[m * 40 + 16 + lane15] = (__bf16)p1;
    }

    // O += P * V  (A = P 16x32 reloaded in A-layout, B = Vt)
    {
      const __bf16* apt = lds_p + (w * 16 + lane15) * 40;
      v16bf a = ld16(apt + half * 16, apt + half * 16 + 8);
#pragma unroll
      for (int nt = 0; nt < 4; ++nt) {
        const __bf16* bpt = lds_vt[cur] + (nt * 16 + lane15) * 40 + half * 16;
        v16bf bb = ld16(bpt, bpt + 8);
        o[nt] = __builtin_amdgcn_wmma_f32_16x16x32_bf16(
            false, a, false, bb, (short)0, o[nt], false, false);
      }
    }

    if (ji + 1 < nj) storeKV(nxt);
#if USE_TDM
    if (w == 0) __builtin_amdgcn_s_wait_tensorcnt(0);
#endif
    __syncthreads();
  }

  // normalize + write O as [B,T,H*D] bf16 (ready for Wo GEMM)
#pragma unroll
  for (int nt = 0; nt < 4; ++nt) {
#pragma unroll
    for (int i = 0; i < 8; ++i) {
      float inv = rsum[i] > 0.f ? 1.f / rsum[i] : 0.f;
      int tq = qbase + w * 16 + i + 8 * half;
      size_t idx = ((size_t)b * TT + tq) * DM + h * HD + nt * 16 + lane15;
      O[idx] = (__bf16)(o[nt][i] * inv);
    }
  }
}

// ---------------------------------------------------------------- launcher
extern "C" void kernel_launch(void* const* d_in, const int* in_sizes, int n_in,
                              void* d_out, int out_size, void* d_ws, size_t ws_size,
                              hipStream_t stream) {
  (void)in_sizes; (void)n_in; (void)out_size; (void)ws_size;

  const float* x     = (const float*)d_in[0];
  // d_in[1] = mask (causal, derived analytically)
  const float* n1w   = (const float*)d_in[2];
  const float* Wqkv  = (const float*)d_in[3];
  const float* bqkv  = (const float*)d_in[4];
  const float* Wo    = (const float*)d_in[5];
  const float* bo    = (const float*)d_in[6];
  const float* n2w   = (const float*)d_in[7];
  const float* Wgate = (const float*)d_in[8];
  const float* Wup   = (const float*)d_in[9];
  const float* Wdown = (const float*)d_in[10];
  float* out = (float*)d_out;

  char* base = (char*)d_ws;
  size_t off = 0;
  auto alloc = [&](size_t bytes) -> void* {
    void* q = base + off;
    off += (bytes + 255) & ~(size_t)255;
    return q;
  };

  __bf16* wqkv_b  = (__bf16*)alloc((size_t)DM * D3 * 2);
  __bf16* wo_b    = (__bf16*)alloc((size_t)DM * DM * 2);
  __bf16* wgate_b = (__bf16*)alloc((size_t)DM * DFF * 2);
  __bf16* wup_b   = (__bf16*)alloc((size_t)DM * DFF * 2);
  __bf16* wdown_b = (__bf16*)alloc((size_t)DFF * DM * 2);
  __bf16* h_b     = (__bf16*)alloc((size_t)BT * DM * 2);
  float*  qkv_f   = (float*) alloc((size_t)BT * D3 * 4);
  __bf16* q_b     = (__bf16*)alloc((size_t)BT * DM * 2);
  __bf16* k_b     = (__bf16*)alloc((size_t)BT * DM * 2);
  __bf16* v_b     = (__bf16*)alloc((size_t)BT * DM * 2);
  __bf16* o_b     = (__bf16*)alloc((size_t)BT * DM * 2);
  float*  x2_f    = (float*) alloc((size_t)BT * DM * 4);
  __bf16* gate_b  = (__bf16*)alloc((size_t)BT * DFF * 2);
  __bf16* up_b    = (__bf16*)alloc((size_t)BT * DFF * 2);

  // weight conversion fp32 -> bf16
  cvt_f32_bf16<<<DM * D3 / 256, 256, 0, stream>>>(Wqkv, wqkv_b, DM * D3);
  cvt_f32_bf16<<<DM * DM / 256, 256, 0, stream>>>(Wo, wo_b, DM * DM);
  cvt_f32_bf16<<<DM * DFF / 256, 256, 0, stream>>>(Wgate, wgate_b, DM * DFF);
  cvt_f32_bf16<<<DM * DFF / 256, 256, 0, stream>>>(Wup, wup_b, DM * DFF);
  cvt_f32_bf16<<<DFF * DM / 256, 256, 0, stream>>>(Wdown, wdown_b, DFF * DM);

  // h = rmsnorm(x)
  rmsnorm_bf16<<<BT, 256, 0, stream>>>(x, n1w, h_b);

  // qkv = h @ Wqkv + bqkv   (fp32 out for rope precision)
  wmma_gemm<true, false, false><<<dim3(D3 / 128, BT / 128), 256, 0, stream>>>(
      h_b, wqkv_b, bqkv, nullptr, qkv_f, D3, DM);

  // rope + split into [B,H,T,D] bf16
  rope_split<<<BT, 256, 0, stream>>>(qkv_f, q_b, k_b, v_b);

  // causal flash attention -> o_b [B,T,C] bf16
  flash_attn<<<dim3(TT / 64, BB * NH), 128, 0, stream>>>(q_b, k_b, v_b, o_b);

  // x2 = x + attn @ Wo + bo
  wmma_gemm<true, true, false><<<dim3(DM / 128, BT / 128), 256, 0, stream>>>(
      o_b, wo_b, bo, x, x2_f, DM, DM);

  // h2 = rmsnorm(x2)
  rmsnorm_bf16<<<BT, 256, 0, stream>>>(x2_f, n2w, h_b);

  // gate / up
  wmma_gemm<false, false, true><<<dim3(DFF / 128, BT / 128), 256, 0, stream>>>(
      h_b, wgate_b, nullptr, nullptr, gate_b, DFF, DM);
  wmma_gemm<false, false, true><<<dim3(DFF / 128, BT / 128), 256, 0, stream>>>(
      h_b, wup_b, nullptr, nullptr, up_b, DFF, DM);

  // mff = silu(gate) * up   (in place into gate_b)
  silu_mul<<<BT * DFF / 1024, 256, 0, stream>>>(gate_b, up_b, gate_b);

  // out = x2 + mff @ Wdown
  wmma_gemm<false, true, false><<<dim3(DM / 128, BT / 128), 256, 0, stream>>>(
      gate_b, wdown_b, nullptr, x2_f, out, DM, DFF);
}
